// MultiHeadAttention_69389491634833
// MI455X (gfx1250) — compile-verified, matrix operand order checked by reading, not run on hardware
//
#include <hip/hip_runtime.h>

// ---------------------------------------------------------------------------
// MHA forward for MI455X (gfx1250, wave32, WMMA f32_16x16x32_f16,
// async global->LDS double-buffered GEMM staging)
//   B=2, S=2048, D_MODEL=1024, H=16, HD=64
// ---------------------------------------------------------------------------

typedef __attribute__((ext_vector_type(16))) _Float16 v16h;
typedef __attribute__((ext_vector_type(8)))  _Float16 v8h;
typedef __attribute__((ext_vector_type(4)))  _Float16 v4h;
typedef __attribute__((ext_vector_type(8)))  float    v8f;
typedef __attribute__((ext_vector_type(4)))  float    v4f;

#define NBATCH   2
#define SEQ      2048
#define DMODEL   1024
#define NHEADS   16
#define HDIM     64

__device__ __forceinline__ v16h cat16(v8h lo, v8h hi) {
  return __builtin_shufflevector(lo, hi, 0,1,2,3,4,5,6,7,8,9,10,11,12,13,14,15);
}

__device__ __forceinline__ v8f wmma_f16(v16h a, v16h b, v8f c) {
  // (neg_a, A, neg_b, B, c_mod, C, reuse_a, reuse_b)
  return __builtin_amdgcn_wmma_f32_16x16x32_f16(false, a, false, b, (short)0, c,
                                                false, false);
}

// CDNA5 async global->LDS copy, 16 bytes per lane, tracked by ASYNCcnt.
__device__ __forceinline__ void async_b128(unsigned lds_off,
                                           const _Float16* gaddr) {
  asm volatile("global_load_async_to_lds_b128 %0, %1, off"
               :: "v"(lds_off), "v"(gaddr) : "memory");
}
__device__ __forceinline__ void wait_async0() {
  asm volatile("s_wait_asynccnt 0" ::: "memory");
}

__device__ __forceinline__ float red_max16(float x) {
  x = fmaxf(x, __shfl_xor(x, 1, 32));
  x = fmaxf(x, __shfl_xor(x, 2, 32));
  x = fmaxf(x, __shfl_xor(x, 4, 32));
  x = fmaxf(x, __shfl_xor(x, 8, 32));
  return x;
}

__device__ __forceinline__ float red_sum16(float x) {
  x += __shfl_xor(x, 1, 32);
  x += __shfl_xor(x, 2, 32);
  x += __shfl_xor(x, 4, 32);
  x += __shfl_xor(x, 8, 32);
  return x;
}

// ---------------------------------------------------------------------------
// Prep kernels: one-time f32 -> f16 conversion (and weight transpose) so the
// GEMM tiles become pure contiguous copies (async-to-LDS friendly) and all
// WMMA fragments are two contiguous 16B loads per lane.
// ---------------------------------------------------------------------------
__global__ __launch_bounds__(256) void cvt_x_kernel(
    const float* __restrict__ x, _Float16* __restrict__ xh) {
  const size_t i = ((size_t)blockIdx.x * 256 + threadIdx.x) * 4;
  v4f f = *(const v4f*)(x + i);
  v4h h;
  h[0] = (_Float16)f[0]; h[1] = (_Float16)f[1];
  h[2] = (_Float16)f[2]; h[3] = (_Float16)f[3];
  *(v4h*)(xh + i) = h;
}

// dst[n][k] (f16) = src[k][n] (f32);  K, N multiples of 32
__global__ __launch_bounds__(256) void transpose_w_kernel(
    const float* __restrict__ src, _Float16* __restrict__ dst, int K, int N) {
  __shared__ float tile[32][33];
  const int k0 = blockIdx.y * 32;
  const int n0 = blockIdx.x * 32;
  const int tx = threadIdx.x;        // 0..31
  const int ty = threadIdx.y;        // 0..7
#pragma unroll
  for (int i = 0; i < 32; i += 8)
    tile[ty + i][tx] = src[(size_t)(k0 + ty + i) * N + n0 + tx];
  __syncthreads();
#pragma unroll
  for (int i = 0; i < 32; i += 8)
    dst[(size_t)(n0 + ty + i) * K + k0 + tx] = (_Float16)tile[tx][ty + i];
}

// ---------------------------------------------------------------------------
// Shared GEMM main loop: acc[4][2] += A[128 rows @ mb] x B[128 cols @ nb],
// K = 1024, both operands f16 with k contiguous (A row-major, B stored [n][k]).
// 256 threads / 8 waves; wave (wm,wn) owns a 64x32 slab = 4x2 WMMA tiles.
// Double-buffered LDS tiles filled with async global->LDS b128 copies.
// ---------------------------------------------------------------------------
__device__ __forceinline__ void gemm_mainloop_k1024(
    const _Float16* __restrict__ A, const _Float16* __restrict__ B,
    int mb, int nb, v8f acc[4][2]) {
  __shared__ _Float16 Ah[2][128][40];   // 80B row stride (16B multiple)
  __shared__ _Float16 Bh[2][128][40];

  const int tid  = threadIdx.x;
  const int lane = tid & 31;
  const int row  = lane & 15;
  const int grp  = lane >> 4;
  const int wave = tid >> 5;
  const int wm   = wave >> 2;          // 0..1
  const int wn   = wave & 3;           // 0..3

  const int sr = tid >> 1;             // staged row 0..127
  const int sc = (tid & 1) * 16;       // staged col offset {0,16}
  const _Float16* ga0 = A + (size_t)(mb + sr) * DMODEL + sc;
  const _Float16* gb0 = B + (size_t)(nb + sr) * DMODEL + sc;

  // issue tile 0
  {
    const unsigned la = (unsigned)(uintptr_t)&Ah[0][sr][sc];
    const unsigned lb = (unsigned)(uintptr_t)&Bh[0][sr][sc];
    async_b128(la,      ga0);
    async_b128(la + 16, ga0 + 8);
    async_b128(lb,      gb0);
    async_b128(lb + 16, gb0 + 8);
  }
  wait_async0();
  __syncthreads();

  int buf = 0;
  for (int kk = 0; kk < DMODEL; kk += 32) {
    const bool more = (kk + 32) < DMODEL;
    if (more) {   // prefetch next tile into the other buffer
      const unsigned la = (unsigned)(uintptr_t)&Ah[buf ^ 1][sr][sc];
      const unsigned lb = (unsigned)(uintptr_t)&Bh[buf ^ 1][sr][sc];
      async_b128(la,      ga0 + kk + 32);
      async_b128(la + 16, ga0 + kk + 40);
      async_b128(lb,      gb0 + kk + 32);
      async_b128(lb + 16, gb0 + kk + 40);
    }

    v16h af[4];
#pragma unroll
    for (int tm = 0; tm < 4; ++tm) {
      const int r = wm * 64 + tm * 16 + row;
      af[tm] = cat16(*(const v8h*)&Ah[buf][r][8 * grp],
                     *(const v8h*)&Ah[buf][r][16 + 8 * grp]);
    }
    v16h bf[2];
#pragma unroll
    for (int tn = 0; tn < 2; ++tn) {
      const int n = wn * 32 + tn * 16 + row;
      bf[tn] = cat16(*(const v8h*)&Bh[buf][n][8 * grp],
                     *(const v8h*)&Bh[buf][n][16 + 8 * grp]);
    }
#pragma unroll
    for (int tm = 0; tm < 4; ++tm)
#pragma unroll
      for (int tn = 0; tn < 2; ++tn)
        acc[tm][tn] = wmma_f16(af[tm], bf[tn], acc[tm][tn]);

    if (more) {
      wait_async0();        // our async copies into buf^1 are done
      __syncthreads();      // everyone finished reading buf / writing buf^1
    }
    buf ^= 1;
  }
}

// ---------------------------------------------------------------------------
// Kernel 1: QKV projection.  C[4096][3072] = xh @ Wqt^T + b_qkv.  Scatters:
//   q_ws: [bh][s][64]  (pre-scaled by 1/sqrt(64))
//   k_ws: [bh][s][64]
//   v_ws: [bh][d][s]   (transposed for the PV matmul's B-fragment layout)
// ---------------------------------------------------------------------------
__global__ __launch_bounds__(256) void qkv_gemm_kernel(
    const _Float16* __restrict__ xh, const _Float16* __restrict__ Wqt,
    const float* __restrict__ bqkv,
    _Float16* __restrict__ q_ws, _Float16* __restrict__ k_ws,
    _Float16* __restrict__ v_ws) {
  const int lane = threadIdx.x & 31;
  const int row  = lane & 15;
  const int grp  = lane >> 4;
  const int wave = threadIdx.x >> 5;
  const int wm   = wave >> 2;
  const int wn   = wave & 3;
  const int mb   = blockIdx.y * 128;
  const int nb   = blockIdx.x * 128;

  v8f acc[4][2];
#pragma unroll
  for (int i = 0; i < 4; ++i)
#pragma unroll
    for (int j = 0; j < 2; ++j) {
      v8f z = {0.f,0.f,0.f,0.f,0.f,0.f,0.f,0.f};
      acc[i][j] = z;
    }

  gemm_mainloop_k1024(xh, Wqt, mb, nb, acc);

#pragma unroll
  for (int tm = 0; tm < 4; ++tm) {
#pragma unroll
    for (int tn = 0; tn < 2; ++tn) {
      const int n    = nb + wn * 32 + tn * 16 + row;   // 0..3071
      const float bb = bqkv[n];
      const int h    = n / (3 * HDIM);
      const int c    = n % (3 * HDIM);
#pragma unroll
      for (int r = 0; r < 8; ++r) {
        const int m  = mb + wm * 64 + tm * 16 + r + 8 * grp;
        const int b  = m >> 11;
        const int s  = m & (SEQ - 1);
        const int bh = b * NHEADS + h;
        const float val = acc[tm][tn][r] + bb;
        if (c < HDIM) {
          q_ws[((size_t)bh * SEQ + s) * HDIM + c] = (_Float16)(val * 0.125f);
        } else if (c < 2 * HDIM) {
          k_ws[((size_t)bh * SEQ + s) * HDIM + (c - HDIM)] = (_Float16)val;
        } else {
          v_ws[((size_t)bh * HDIM + (c - 2 * HDIM)) * SEQ + s] = (_Float16)val;
        }
      }
    }
  }
}

// ---------------------------------------------------------------------------
// Kernel 2: flash-style attention.  One wave per 16-query tile; 32 keys per
// iteration.  All WMMA fragments loaded directly from global (L2-resident).
// ---------------------------------------------------------------------------
__global__ __launch_bounds__(128) void attn_kernel(
    const _Float16* __restrict__ q_ws, const _Float16* __restrict__ k_ws,
    const _Float16* __restrict__ v_ws, const int* __restrict__ mask,
    _Float16* __restrict__ att_ws) {
  __shared__ _Float16 plds[4][16 * 32];   // per-wave P transpose scratch

  const int lane = threadIdx.x & 31;
  const int row  = lane & 15;
  const int grp  = lane >> 4;
  const int wave = threadIdx.x >> 5;

  const int bh = blockIdx.x >> 5;                       // 0..31
  const int qt = (blockIdx.x & 31) * 4 + wave;          // 0..127
  const int q0 = qt * 16;
  const int b  = bh >> 4;
  const int hh = bh & (NHEADS - 1);

  const _Float16* qb = q_ws + (size_t)bh * SEQ * HDIM;
  const _Float16* kb = k_ws + (size_t)bh * SEQ * HDIM;
  const _Float16* vb = v_ws + (size_t)bh * HDIM * SEQ;
  const int* mrow    = mask + b * SEQ;

  const _Float16* qrow = qb + (size_t)(q0 + row) * HDIM;
  const v16h aq0 = cat16(*(const v8h*)(qrow + 8 * grp),
                         *(const v8h*)(qrow + 16 + 8 * grp));
  const v16h aq1 = cat16(*(const v8h*)(qrow + 32 + 8 * grp),
                         *(const v8h*)(qrow + 48 + 8 * grp));

  float mv[8], lv[8];
  v8f o[4];
#pragma unroll
  for (int t = 0; t < 4; ++t) {
    v8f z = {0.f,0.f,0.f,0.f,0.f,0.f,0.f,0.f};
    o[t] = z;
  }
#pragma unroll
  for (int r = 0; r < 8; ++r) { mv[r] = -__builtin_inff(); lv[r] = 0.f; }

  _Float16* pl = plds[wave];

  for (int s0 = 0; s0 < SEQ; s0 += 32) {
    const _Float16* k0r = kb + (size_t)(s0 + row) * HDIM;
    const _Float16* k1r = kb + (size_t)(s0 + 16 + row) * HDIM;
    const v16h bk00 = cat16(*(const v8h*)(k0r + 8 * grp),
                            *(const v8h*)(k0r + 16 + 8 * grp));
    const v16h bk01 = cat16(*(const v8h*)(k0r + 32 + 8 * grp),
                            *(const v8h*)(k0r + 48 + 8 * grp));
    const v16h bk10 = cat16(*(const v8h*)(k1r + 8 * grp),
                            *(const v8h*)(k1r + 16 + 8 * grp));
    const v16h bk11 = cat16(*(const v8h*)(k1r + 32 + 8 * grp),
                            *(const v8h*)(k1r + 48 + 8 * grp));

    v8f c0 = {0.f,0.f,0.f,0.f,0.f,0.f,0.f,0.f};
    v8f c1 = {0.f,0.f,0.f,0.f,0.f,0.f,0.f,0.f};
    c0 = wmma_f16(aq0, bk00, c0);
    c0 = wmma_f16(aq1, bk01, c0);
    c1 = wmma_f16(aq0, bk10, c1);
    c1 = wmma_f16(aq1, bk11, c1);

    const bool dead0 = (mrow[s0 + row] == 0);
    const bool dead1 = (mrow[s0 + 16 + row] == 0);
#pragma unroll
    for (int r = 0; r < 8; ++r) {
      if (dead0) c0[r] = -__builtin_inff();
      if (dead1) c1[r] = -__builtin_inff();
    }

    float p0[8], p1[8], alpha[8];
#pragma unroll
    for (int r = 0; r < 8; ++r) {
      float rm = red_max16(fmaxf(c0[r], c1[r]));
      const float mn = fmaxf(mv[r], rm);
      alpha[r] = __expf(mv[r] - mn);
      mv[r]    = mn;
      p0[r] = __expf(c0[r] - mn);
      p1[r] = __expf(c1[r] - mn);
      lv[r] = lv[r] * alpha[r] + red_sum16(p0[r] + p1[r]);
    }
#pragma unroll
    for (int t = 0; t < 4; ++t)
#pragma unroll
      for (int r = 0; r < 8; ++r) o[t][r] *= alpha[r];

    // P (C-tile layout) -> LDS -> A-fragment layout (per-wave, DS in-order)
#pragma unroll
    for (int r = 0; r < 8; ++r) {
      const int pr = r + 8 * grp;
      pl[pr * 32 + row]      = (_Float16)p0[r];
      pl[pr * 32 + 16 + row] = (_Float16)p1[r];
    }
    asm volatile("s_wait_dscnt 0" ::: "memory");
    const v16h pa = cat16(*(const v8h*)&pl[row * 32 + 8 * grp],
                          *(const v8h*)&pl[row * 32 + 16 + 8 * grp]);

#pragma unroll
    for (int t = 0; t < 4; ++t) {
      const _Float16* vr = vb + (size_t)(t * 16 + row) * SEQ + s0;
      const v16h bv = cat16(*(const v8h*)(vr + 8 * grp),
                            *(const v8h*)(vr + 16 + 8 * grp));
      o[t] = wmma_f16(pa, bv, o[t]);
    }
  }

#pragma unroll
  for (int t = 0; t < 4; ++t) {
#pragma unroll
    for (int r = 0; r < 8; ++r) {
      const int q = q0 + r + 8 * grp;
      const int d = t * 16 + row;
      const float val = o[t][r] / lv[r];
      att_ws[((size_t)(b * SEQ + q)) * DMODEL + hh * HDIM + d] = (_Float16)val;
    }
  }
}

// ---------------------------------------------------------------------------
// Kernel 3: output projection.  out[4096][1024] = att @ Wot^T + b_out
// ---------------------------------------------------------------------------
__global__ __launch_bounds__(256) void out_gemm_kernel(
    const _Float16* __restrict__ att, const _Float16* __restrict__ Wot,
    const float* __restrict__ bout, float* __restrict__ out) {
  const int lane = threadIdx.x & 31;
  const int row  = lane & 15;
  const int grp  = lane >> 4;
  const int wave = threadIdx.x >> 5;
  const int wm   = wave >> 2;
  const int wn   = wave & 3;
  const int mb   = blockIdx.y * 128;
  const int nb   = blockIdx.x * 128;

  v8f acc[4][2];
#pragma unroll
  for (int i = 0; i < 4; ++i)
#pragma unroll
    for (int j = 0; j < 2; ++j) {
      v8f z = {0.f,0.f,0.f,0.f,0.f,0.f,0.f,0.f};
      acc[i][j] = z;
    }

  gemm_mainloop_k1024(att, Wot, mb, nb, acc);

#pragma unroll
  for (int tm = 0; tm < 4; ++tm) {
#pragma unroll
    for (int tn = 0; tn < 2; ++tn) {
      const int n    = nb + wn * 32 + tn * 16 + row;
      const float bb = bout[n];
#pragma unroll
      for (int r = 0; r < 8; ++r) {
        const int m = mb + wm * 64 + tm * 16 + r + 8 * grp;
        out[(size_t)m * DMODEL + n] = acc[tm][tn][r] + bb;
      }
    }
  }
}

// ---------------------------------------------------------------------------
// Launch
// ---------------------------------------------------------------------------
extern "C" void kernel_launch(void* const* d_in, const int* in_sizes, int n_in,
                              void* d_out, int out_size, void* d_ws,
                              size_t ws_size, hipStream_t stream) {
  const float* x    = (const float*)d_in[0];
  const int*   mask = (const int*)d_in[1];
  const float* Wqkv = (const float*)d_in[2];
  const float* bqkv = (const float*)d_in[3];
  const float* Wout = (const float*)d_in[4];
  const float* bout = (const float*)d_in[5];
  float* out = (float*)d_out;

  const size_t nBS  = (size_t)NBATCH * SEQ;            // 4096
  _Float16* xh     = (_Float16*)d_ws;                  // [4096][1024]
  _Float16* Wqt    = xh  + nBS * DMODEL;               // [3072][1024]
  _Float16* Wot    = Wqt + (size_t)3 * DMODEL * DMODEL;// [1024][1024]
  _Float16* q_ws   = Wot + (size_t)DMODEL * DMODEL;    // [32][2048][64]
  _Float16* k_ws   = q_ws + nBS * DMODEL;
  _Float16* v_ws   = k_ws + nBS * DMODEL;
  _Float16* att_ws = v_ws + nBS * DMODEL;              // [4096][1024]

  // Prep: f16 conversion + weight transposes (one-time, bandwidth-trivial)
  cvt_x_kernel<<<(unsigned)(nBS * DMODEL / 1024), 256, 0, stream>>>(x, xh);
  transpose_w_kernel<<<dim3(3 * DMODEL / 32, DMODEL / 32), dim3(32, 8), 0,
                       stream>>>(Wqkv, Wqt, DMODEL, 3 * DMODEL);
  transpose_w_kernel<<<dim3(DMODEL / 32, DMODEL / 32), dim3(32, 8), 0,
                       stream>>>(Wout, Wot, DMODEL, DMODEL);

  // QKV projection: M=4096, N=3072
  qkv_gemm_kernel<<<dim3(3 * DMODEL / 128, (unsigned)(nBS / 128)), 256, 0,
                    stream>>>(xh, Wqt, bqkv, q_ws, k_ws, v_ws);

  // Attention: 32 (b,h) pairs x 32 query blocks of 64; 4 waves/block
  attn_kernel<<<NBATCH * NHEADS * (SEQ / 64), 128, 0, stream>>>(
      q_ws, k_ws, v_ws, mask, att_ws);

  // Output projection: M=4096, N=1024
  out_gemm_kernel<<<dim3(DMODEL / 128, (unsigned)(nBS / 128)), 256, 0,
                    stream>>>(att_ws, Wot, bout, out);
}
